// _DAHead_39092792329029
// MI455X (gfx1250) — compile-verified, matrix operand order checked by reading, not run on hardware
//
#include <hip/hip_runtime.h>
#include <hip/hip_bf16.h>
#include <math.h>

typedef _Float16 half_t;
typedef __attribute__((ext_vector_type(16))) _Float16 v16h;
typedef __attribute__((ext_vector_type(8)))  _Float16 v8h;
typedef __attribute__((ext_vector_type(8)))  float    v8f;

#define EPSV 1e-5f
#define CIN 512
#define CI  64
#define HWP 4096
#define NB  4
#define KSPLIT 8

__device__ __forceinline__ v8f wmma_f16(v16h a, v16h b, v8f c) {
  return __builtin_amdgcn_wmma_f32_16x16x32_f16(false, a, false, b, (short)0, c, false, false);
}

// Load a 16x32 f16 fragment in the CDNA5 A-matrix layout from a K-contiguous
// [16 rows][ld] buffer.  Works for both A ([M][K]) and B-transposed ([N][K])
// operands.  Lane<16 holds K {kb..kb+7, kb+16..kb+23} with kb=0; lane>=16 kb=8.
__device__ __forceinline__ v16h load_frag(const half_t* p, int ld) {
  const int lane = threadIdx.x & 31;
  const int r    = lane & 15;
  const int kb   = (lane & 16) ? 8 : 0;
  const half_t* q = p + (size_t)r * ld + kb;
  v16h out;
  reinterpret_cast<v8h*>(&out)[0] = *reinterpret_cast<const v8h*>(q);
  reinterpret_cast<v8h*>(&out)[1] = *reinterpret_cast<const v8h*>(q + 16);
  return out;
}

// ---------------- K0: fold BN into weights, convert to f16 -----------------
__global__ void k_fold(const float* wp1, const float* gp1, const float* bp1, const float* mp1, const float* vp1,
                       const float* wc1, const float* gc1, const float* bc1, const float* mc1, const float* vc1,
                       const float* wb,  const float* bb,  const float* wc2, const float* bc2,
                       const float* wd,  const float* bd,
                       const float* wo,  const float* go,  const float* bo,  const float* mo, const float* vo,
                       half_t* Wcat, float* bcat,
                       half_t* Wq, half_t* Wk, half_t* Wv, float* bq, float* bk, float* bv,
                       half_t* Wo, float* boF) {
  int gid = blockIdx.x * blockDim.x + threadIdx.x;
  int gstride = gridDim.x * blockDim.x;
  for (int idx = gid; idx < 128 * CIN; idx += gstride) {
    int o = idx >> 9, c = idx & (CIN - 1);
    float w, inv;
    if (o < CI) { inv = gp1[o] * rsqrtf(vp1[o] + EPSV); w = wp1[o * CIN + c]; }
    else { int oo = o - CI; inv = gc1[oo] * rsqrtf(vc1[oo] + EPSV); w = wc1[oo * CIN + c]; }
    Wcat[idx] = (half_t)(w * inv);
  }
  if (gid < 128) {
    float inv, bias;
    if (gid < CI) { inv = gp1[gid] * rsqrtf(vp1[gid] + EPSV); bias = bp1[gid] - mp1[gid] * inv; }
    else { int oo = gid - CI; inv = gc1[oo] * rsqrtf(vc1[oo] + EPSV); bias = bc1[oo] - mc1[oo] * inv; }
    bcat[gid] = bias;
  }
  for (int idx = gid; idx < CI * CI; idx += gstride) {
    Wq[idx] = (half_t)wb[idx];
    Wk[idx] = (half_t)wc2[idx];
    Wv[idx] = (half_t)wd[idx];
  }
  if (gid < CI) { bq[gid] = bb[gid]; bk[gid] = bc2[gid]; bv[gid] = bd[gid]; }
  for (int idx = gid; idx < CIN * CI; idx += gstride) {
    int o = idx / CI;
    float inv = go[o] * rsqrtf(vo[o] + EPSV);
    Wo[idx] = (half_t)(wo[idx] * inv);
  }
  for (int o = gid; o < CIN; o += gstride) {
    float inv = go[o] * rsqrtf(vo[o] + EPSV);
    boF[o] = bo[o] - mo[o] * inv;
  }
}

// ---------------- K1: x [B][512][4096] f32 -> xT [B][4096][512] f16 --------
__global__ void k_transpose_x(const float* x, half_t* xT) {
  __shared__ float tile[32][33];
  int p0 = blockIdx.x * 32, c0 = blockIdx.y * 32, b = blockIdx.z;
  const float* xb = x + (size_t)b * CIN * HWP;
  half_t* xTb = xT + (size_t)b * HWP * CIN;
  int tx = threadIdx.x, ty = threadIdx.y;
#pragma unroll
  for (int i = 0; i < 4; ++i) {
    int cc = ty * 4 + i;
    tile[cc][tx] = xb[(size_t)(c0 + cc) * HWP + p0 + tx];
  }
  __syncthreads();
#pragma unroll
  for (int i = 0; i < 4; ++i) {
    int pp = ty * 4 + i;
    xTb[(size_t)(p0 + pp) * CIN + c0 + tx] = (half_t)tile[tx][pp];
  }
}

// ---------------- K2: pT[b][pix][128] = xT * Wcat^T + bcat (BN folded) -----
__global__ void k_gemm_in(const half_t* xT, const half_t* Wcat, const float* bcat, half_t* pT) {
  int mt = blockIdx.x, b = blockIdx.y;
  int wv = threadIdx.x >> 5, lane = threadIdx.x & 31;
  const half_t* A = xT + (size_t)b * HWP * CIN + (size_t)mt * 16 * CIN;
  const half_t* B = Wcat + (size_t)wv * 16 * CIN;
  float cb = bcat[wv * 16 + (lane & 15)];
  v8f acc;
#pragma unroll
  for (int r = 0; r < 8; ++r) acc[r] = cb;
  for (int kk = 0; kk < CIN; kk += 32)
    acc = wmma_f16(load_frag(A + kk, CIN), load_frag(B + kk, CIN), acc);
  half_t* out = pT + (size_t)b * HWP * 128;
  int n = wv * 16 + (lane & 15);
  int m0 = mt * 16 + ((lane & 16) ? 8 : 0);
#pragma unroll
  for (int r = 0; r < 8; ++r)
    out[(size_t)(m0 + r) * 128 + n] = (half_t)acc[r];
}

// ---------------- K3: fcb_cm[b][64][4096] = transpose(pT[:, 64:128]) -------
__global__ void k_transpose_fcb(const half_t* pT, half_t* fcb_cm) {
  __shared__ half_t tile[32][33];
  int p0 = blockIdx.x * 32, c0 = blockIdx.y * 32, b = blockIdx.z;
  const half_t* pb = pT + (size_t)b * HWP * 128;
  half_t* fb = fcb_cm + (size_t)b * CI * HWP;
  int tx = threadIdx.x, ty = threadIdx.y;
#pragma unroll
  for (int i = 0; i < 4; ++i) {
    int pp = ty * 4 + i;
    tile[pp][tx] = pb[(size_t)(p0 + pp) * 128 + 64 + c0 + tx];
  }
  __syncthreads();
#pragma unroll
  for (int i = 0; i < 4; ++i) {
    int cc = ty * 4 + i;
    fb[(size_t)(c0 + cc) * HWP + p0 + tx] = tile[tx][cc];
  }
}

// ---------------- K4: q/k (pixel-major) and v (channel-major) projections --
__global__ void k_gemm_qkv(const half_t* pT, const half_t* Wq, const half_t* Wk, const half_t* Wv,
                           const float* bq, const float* bk, const float* bv,
                           half_t* fbT, half_t* fcT, half_t* fd_cm) {
  int pt = blockIdx.x, op = blockIdx.y, b = blockIdx.z;
  int wv = threadIdx.x >> 5, lane = threadIdx.x & 31;
  const half_t* pTb = pT + (size_t)b * HWP * 128;
  if (op < 2) {
    const half_t* W = (op == 0) ? Wq : Wk;
    const float* bias = (op == 0) ? bq : bk;
    half_t* out = ((op == 0) ? fbT : fcT) + (size_t)b * HWP * CI;
    const half_t* A = pTb + (size_t)pt * 16 * 128;   // fp part, ld=128
    const half_t* B = W + (size_t)wv * 16 * CI;
    float cb = bias[wv * 16 + (lane & 15)];
    v8f acc;
#pragma unroll
    for (int r = 0; r < 8; ++r) acc[r] = cb;
    acc = wmma_f16(load_frag(A, 128),      load_frag(B, CI),      acc);
    acc = wmma_f16(load_frag(A + 32, 128), load_frag(B + 32, CI), acc);
    int n = wv * 16 + (lane & 15);
    int m0 = pt * 16 + ((lane & 16) ? 8 : 0);
#pragma unroll
    for (int r = 0; r < 8; ++r)
      out[(size_t)(m0 + r) * CI + n] = (half_t)acc[r];
  } else {  // fd channel-major: C[ch][pix] = Wv * fp
    const half_t* A = Wv + (size_t)wv * 16 * CI;
    const half_t* B = pTb + (size_t)pt * 16 * 128;
    int hi = (lane & 16) ? 8 : 0;
    v8f acc;
#pragma unroll
    for (int r = 0; r < 8; ++r) acc[r] = bv[wv * 16 + hi + r];
    acc = wmma_f16(load_frag(A, CI),      load_frag(B, 128),      acc);
    acc = wmma_f16(load_frag(A + 32, CI), load_frag(B + 32, 128), acc);
    half_t* out = fd_cm + (size_t)b * CI * HWP;
    int n = pt * 16 + (lane & 15);
#pragma unroll
    for (int r = 0; r < 8; ++r)
      out[(size_t)(wv * 16 + hi + r) * HWP + n] = (half_t)acc[r];
  }
}

// ---------------- K5a: channel gram, K split 8 ways (deterministic) --------
__global__ void k_gram_part(const half_t* fcb_cm, float* attn_part) {
  int b = blockIdx.x, ks = blockIdx.y;
  int wv = threadIdx.x >> 5, lane = threadIdx.x & 31;
  const half_t* fa = fcb_cm + (size_t)b * CI * HWP + (size_t)ks * (HWP / KSPLIT);
  float* outb = attn_part + ((size_t)ks * NB + b) * CI * CI;
  for (int t = wv; t < 16; t += 8) {
    int ct = t >> 2, dt = t & 3;
    const half_t* A = fa + (size_t)ct * 16 * HWP;
    const half_t* B = fa + (size_t)dt * 16 * HWP;
    v8f acc;
#pragma unroll
    for (int r = 0; r < 8; ++r) acc[r] = 0.f;
    for (int kk = 0; kk < HWP / KSPLIT; kk += 32)
      acc = wmma_f16(load_frag(A + kk, HWP), load_frag(B + kk, HWP), acc);
    int n = dt * 16 + (lane & 15);
    int m0 = ct * 16 + ((lane & 16) ? 8 : 0);
#pragma unroll
    for (int r = 0; r < 8; ++r)
      outb[(size_t)(m0 + r) * CI + n] = acc[r];
  }
}

__global__ void k_gram_reduce(const float* attn_part, float* attn_raw) {
  int idx = blockIdx.x * blockDim.x + threadIdx.x;   // NB*CI*CI total
  float s = 0.f;
#pragma unroll
  for (int ks = 0; ks < KSPLIT; ++ks)
    s += attn_part[(size_t)ks * NB * CI * CI + idx];
  attn_raw[idx] = s;
}

// ---------------- K6: channel softmax of (rowmax - a) ----------------------
__global__ void k_chan_softmax(const float* attn_raw, half_t* attn_h) {
  int row = blockIdx.x, b = blockIdx.y, lane = threadIdx.x;
  const float* a = attn_raw + ((size_t)b * CI + row) * CI;
  float a0 = a[lane], a1 = a[lane + 32];
  float mx = fmaxf(a0, a1);
#pragma unroll
  for (int off = 16; off > 0; off >>= 1) mx = fmaxf(mx, __shfl_xor(mx, off, 32));
  float t0 = mx - a0, t1 = mx - a1;
  float m2 = fmaxf(t0, t1);
#pragma unroll
  for (int off = 16; off > 0; off >>= 1) m2 = fmaxf(m2, __shfl_xor(m2, off, 32));
  float e0 = __expf(t0 - m2), e1 = __expf(t1 - m2);
  float s = e0 + e1;
#pragma unroll
  for (int off = 16; off > 0; off >>= 1) s += __shfl_xor(s, off, 32);
  float inv = 1.f / s;
  half_t* o = attn_h + ((size_t)b * CI + row) * CI;
  o[lane] = (half_t)(e0 * inv);
  o[lane + 32] = (half_t)(e1 * inv);
}

// ---------------- K7: fusionT = fp + fcb + beta * (attn_c @ fa) ------------
__global__ void k_featc(const half_t* pT, const half_t* attn_h, const float* beta, half_t* fusionT) {
  int pt = blockIdx.x, b = blockIdx.y;
  int wv = threadIdx.x >> 5, lane = threadIdx.x & 31;
  float bet = beta[0];
  const half_t* pTb = pT + (size_t)b * HWP * 128;
  const half_t* A = pTb + (size_t)pt * 16 * 128 + 64;            // fcb part
  const half_t* B = attn_h + (size_t)b * CI * CI + (size_t)wv * 16 * CI;
  v8f acc;
#pragma unroll
  for (int r = 0; r < 8; ++r) acc[r] = 0.f;
  acc = wmma_f16(load_frag(A, 128),      load_frag(B, CI),      acc);
  acc = wmma_f16(load_frag(A + 32, 128), load_frag(B + 32, CI), acc);
  half_t* fo = fusionT + (size_t)b * HWP * CI;
  int n = wv * 16 + (lane & 15);
  int m0 = pt * 16 + ((lane & 16) ? 8 : 0);
#pragma unroll
  for (int r = 0; r < 8; ++r) {
    const half_t* src = pTb + (size_t)(m0 + r) * 128;
    float v = bet * acc[r] + (float)src[n] + (float)src[64 + n];  // fp + fcb
    fo[(size_t)(m0 + r) * CI + n] = (half_t)v;
  }
}

// ---------------- K8: fused flash attention, 2 query tiles per wave --------
// K/V fragment loads are shared across both query tiles: 16 b128 loads per
// 16 WMMAs (1:1), vs 1:2 with a single tile.
__global__ void __launch_bounds__(256, 1)
k_attn(const half_t* fbT, const half_t* fcT, const half_t* fd_cm,
       const float* alpha, half_t* fusionT) {
  __shared__ __attribute__((aligned(16))) half_t ldsP[8][2][16][40];
  int b = blockIdx.y;
  int wv = threadIdx.x >> 5, lane = threadIdx.x & 31;
  int qt0 = (blockIdx.x * 8 + wv) * 2;          // two consecutive 16-query tiles
  const half_t* fbTb = fbT + (size_t)b * HWP * CI;
  const half_t* fcTb = fcT + (size_t)b * HWP * CI;
  const half_t* fdb  = fd_cm + (size_t)b * CI * HWP;
  v16h aq[2][2];
#pragma unroll
  for (int t = 0; t < 2; ++t) {
    aq[t][0] = load_frag(fbTb + (size_t)(qt0 + t) * 16 * CI, CI);
    aq[t][1] = load_frag(fbTb + (size_t)(qt0 + t) * 16 * CI + 32, CI);
  }
  v8f o[2][4];
  float mrow[2][8], lrow[2][8];
#pragma unroll
  for (int t = 0; t < 2; ++t)
#pragma unroll
    for (int r = 0; r < 8; ++r) {
      o[t][0][r] = 0.f; o[t][1][r] = 0.f; o[t][2][r] = 0.f; o[t][3][r] = 0.f;
      mrow[t][r] = -1e30f; lrow[t][r] = 0.f;
    }
  const int hi = (lane & 16) ? 8 : 0;
  const int nl = lane & 15;
  for (int k0 = 0; k0 < HWP; k0 += 32) {
    // prefetch next chunk of keys/values into cache while this chunk computes
    __builtin_prefetch(fcTb + (size_t)(k0 + 32) * CI, 0, 3);
    __builtin_prefetch(fdb + (size_t)k0 + 32, 0, 3);
    const half_t* kbp0 = fcTb + (size_t)k0 * CI;
    const half_t* kbp1 = fcTb + (size_t)(k0 + 16) * CI;
    v16h kf00 = load_frag(kbp0, CI);
    v16h kf01 = load_frag(kbp0 + 32, CI);
    v16h kf10 = load_frag(kbp1, CI);
    v16h kf11 = load_frag(kbp1 + 32, CI);
#pragma unroll
    for (int t = 0; t < 2; ++t) {
      v8f s0, s1;
#pragma unroll
      for (int r = 0; r < 8; ++r) { s0[r] = 0.f; s1[r] = 0.f; }
      s0 = wmma_f16(aq[t][0], kf00, s0);
      s0 = wmma_f16(aq[t][1], kf01, s0);
      s1 = wmma_f16(aq[t][0], kf10, s1);
      s1 = wmma_f16(aq[t][1], kf11, s1);
      half_t* pls = &ldsP[wv][t][0][0];
#pragma unroll
      for (int r = 0; r < 8; ++r) {
        float c = fmaxf(s0[r], s1[r]);
        c = fmaxf(c, __shfl_xor(c, 1, 32));
        c = fmaxf(c, __shfl_xor(c, 2, 32));
        c = fmaxf(c, __shfl_xor(c, 4, 32));
        c = fmaxf(c, __shfl_xor(c, 8, 32));
        float nm = fmaxf(mrow[t][r], c);
        float sc = __expf(mrow[t][r] - nm);
        mrow[t][r] = nm;
        lrow[t][r] *= sc;
        o[t][0][r] *= sc; o[t][1][r] *= sc; o[t][2][r] *= sc; o[t][3][r] *= sc;
        float p0 = __expf(s0[r] - nm), p1 = __expf(s1[r] - nm);
        float ps = p0 + p1;
        ps += __shfl_xor(ps, 1, 32);
        ps += __shfl_xor(ps, 2, 32);
        ps += __shfl_xor(ps, 4, 32);
        ps += __shfl_xor(ps, 8, 32);
        lrow[t][r] += ps;
        pls[(hi + r) * 40 + nl]      = (half_t)p0;   // C-layout -> A-layout via LDS
        pls[(hi + r) * 40 + 16 + nl] = (half_t)p1;
      }
    }
    v16h vf0 = load_frag(fdb + (size_t)0 * 16 * HWP + k0, HWP);
    v16h vf1 = load_frag(fdb + (size_t)1 * 16 * HWP + k0, HWP);
    v16h vf2 = load_frag(fdb + (size_t)2 * 16 * HWP + k0, HWP);
    v16h vf3 = load_frag(fdb + (size_t)3 * 16 * HWP + k0, HWP);
#pragma unroll
    for (int t = 0; t < 2; ++t) {
      v16h pf = load_frag(&ldsP[wv][t][0][0], 40);
      o[t][0] = wmma_f16(pf, vf0, o[t][0]);
      o[t][1] = wmma_f16(pf, vf1, o[t][1]);
      o[t][2] = wmma_f16(pf, vf2, o[t][2]);
      o[t][3] = wmma_f16(pf, vf3, o[t][3]);
    }
  }
  float al = alpha[0];
  half_t* fo = fusionT + (size_t)b * HWP * CI;
#pragma unroll
  for (int t = 0; t < 2; ++t)
#pragma unroll
    for (int r = 0; r < 8; ++r) {
      float rinv = 1.f / lrow[t][r];
      size_t ro = (size_t)((qt0 + t) * 16 + hi + r) * CI;
      fo[ro + 0  + nl] = (half_t)((float)fo[ro + 0  + nl] + al * o[t][0][r] * rinv);
      fo[ro + 16 + nl] = (half_t)((float)fo[ro + 16 + nl] + al * o[t][1][r] * rinv);
      fo[ro + 32 + nl] = (half_t)((float)fo[ro + 32 + nl] + al * o[t][2][r] * rinv);
      fo[ro + 48 + nl] = (half_t)((float)fo[ro + 48 + nl] + al * o[t][3][r] * rinv);
    }
}

// ---------------- K9: out[b][512][4096] = BN(Wo * fusion) fp32 -------------
__global__ void k_out(const half_t* fusionT, const half_t* Wo, const float* boF, float* out) {
  int nt = blockIdx.x, mg = blockIdx.y, b = blockIdx.z;
  int wv = threadIdx.x >> 5, lane = threadIdx.x & 31;
  int mt = mg * 8 + wv;
  const half_t* A = Wo + (size_t)mt * 16 * CI;
  const half_t* B = fusionT + (size_t)b * HWP * CI + (size_t)nt * 16 * CI;
  int hi = (lane & 16) ? 8 : 0;
  v8f acc;
#pragma unroll
  for (int r = 0; r < 8; ++r) acc[r] = boF[mt * 16 + hi + r];
  acc = wmma_f16(load_frag(A, CI),      load_frag(B, CI),      acc);
  acc = wmma_f16(load_frag(A + 32, CI), load_frag(B + 32, CI), acc);
  float* ob = out + (size_t)b * CIN * HWP;
  int n = nt * 16 + (lane & 15);
#pragma unroll
  for (int r = 0; r < 8; ++r)
    ob[(size_t)(mt * 16 + hi + r) * HWP + n] = acc[r];
}

extern "C" void kernel_launch(void* const* d_in, const int* in_sizes, int n_in,
                              void* d_out, int out_size, void* d_ws, size_t ws_size,
                              hipStream_t stream) {
  (void)in_sizes; (void)n_in; (void)out_size; (void)ws_size;
  const float* x    = (const float*)d_in[0];
  const float* wp1  = (const float*)d_in[1];
  const float* gp1  = (const float*)d_in[2];
  const float* bp1  = (const float*)d_in[3];
  const float* mp1  = (const float*)d_in[4];
  const float* vp1  = (const float*)d_in[5];
  const float* wc1  = (const float*)d_in[6];
  const float* gc1  = (const float*)d_in[7];
  const float* bc1  = (const float*)d_in[8];
  const float* mc1  = (const float*)d_in[9];
  const float* vc1  = (const float*)d_in[10];
  const float* wb   = (const float*)d_in[11];
  const float* bb   = (const float*)d_in[12];
  const float* wc2  = (const float*)d_in[13];
  const float* bc2  = (const float*)d_in[14];
  const float* wd   = (const float*)d_in[15];
  const float* bd   = (const float*)d_in[16];
  const float* alpha= (const float*)d_in[17];
  const float* beta = (const float*)d_in[18];
  const float* wo   = (const float*)d_in[19];
  const float* go   = (const float*)d_in[20];
  const float* bo   = (const float*)d_in[21];
  const float* mo   = (const float*)d_in[22];
  const float* vo   = (const float*)d_in[23];
  float* out = (float*)d_out;

  char* ws = (char*)d_ws;
  size_t off = 0;
  auto carve = [&](size_t bytes) -> void* {
    void* p = (void*)(ws + off);
    off += (bytes + 255) & ~(size_t)255;
    return p;
  };
  half_t* xT     = (half_t*)carve((size_t)NB * HWP * CIN * 2);   // 16 MB (reused below)
  half_t* pT     = (half_t*)carve((size_t)NB * HWP * 128 * 2);   // 4 MB
  half_t* fcb_cm = (half_t*)carve((size_t)NB * CI * HWP * 2);    // 2 MB
  half_t* Wcat   = (half_t*)carve(128 * CIN * 2);
  float*  bcat   = (float*)carve(128 * 4);
  half_t* Wq     = (half_t*)carve(CI * CI * 2);
  half_t* Wk     = (half_t*)carve(CI * CI * 2);
  half_t* Wv     = (half_t*)carve(CI * CI * 2);
  float*  bq     = (float*)carve(CI * 4);
  float*  bk     = (float*)carve(CI * 4);
  float*  bv     = (float*)carve(CI * 4);
  half_t* WoH    = (half_t*)carve(CIN * CI * 2);
  float*  boF    = (float*)carve(CIN * 4);
  // xT is dead after k_gemm_in -> alias attention/channel buffers onto it.
  char* ra = (char*)xT;
  half_t* fbT      = (half_t*)(ra);
  half_t* fcT      = (half_t*)(ra + (size_t)1 * NB * HWP * CI * 2);
  half_t* fd_cm    = (half_t*)(ra + (size_t)2 * NB * HWP * CI * 2);
  half_t* fusionT  = (half_t*)(ra + (size_t)3 * NB * HWP * CI * 2);
  float*  attn_raw = (float*) (ra + (size_t)4 * NB * HWP * CI * 2);
  half_t* attn_h   = (half_t*)(ra + (size_t)4 * NB * HWP * CI * 2 + (size_t)NB * CI * CI * 4);
  float*  attn_part= (float*) (ra + (size_t)4 * NB * HWP * CI * 2 + (size_t)NB * CI * CI * 8);

  k_fold<<<32, 256, 0, stream>>>(wp1, gp1, bp1, mp1, vp1, wc1, gc1, bc1, mc1, vc1,
                                 wb, bb, wc2, bc2, wd, bd, wo, go, bo, mo, vo,
                                 Wcat, bcat, Wq, Wk, Wv, bq, bk, bv, WoH, boF);
  k_transpose_x<<<dim3(HWP / 32, CIN / 32, NB), dim3(32, 8), 0, stream>>>(x, xT);
  k_gemm_in<<<dim3(HWP / 16, NB), 256, 0, stream>>>(xT, Wcat, bcat, pT);
  k_transpose_fcb<<<dim3(HWP / 32, CI / 32, NB), dim3(32, 8), 0, stream>>>(pT, fcb_cm);
  k_gemm_qkv<<<dim3(HWP / 16, 3, NB), 128, 0, stream>>>(pT, Wq, Wk, Wv, bq, bk, bv, fbT, fcT, fd_cm);
  k_gram_part<<<dim3(NB, KSPLIT), 256, 0, stream>>>(fcb_cm, attn_part);
  k_gram_reduce<<<dim3(NB * CI * CI / 256), 256, 0, stream>>>(attn_part, attn_raw);
  k_chan_softmax<<<dim3(CI, NB), 32, 0, stream>>>(attn_raw, attn_h);
  k_featc<<<dim3(HWP / 16, NB), 128, 0, stream>>>(pT, attn_h, beta, fusionT);
  k_attn<<<dim3(HWP / (16 * 2 * 8), NB), 256, 0, stream>>>(fbT, fcT, fd_cm, alpha, fusionT);
  k_out<<<dim3(HWP / 16, 4, NB), 256, 0, stream>>>(fusionT, WoH, boF, out);
}
